// ScaNN_55757265437294
// MI455X (gfx1250) — matrix operand order repaired and checked by reading, not compiled, not run on hardware
//
#include <hip/hip_runtime.h>
#include <hip/hip_bf16.h>

// ---------------------------------------------------------------------------
// ScaNN-style exact retrieval with exclusions for MI455X (gfx1250, wave32).
//
// Strategy (memory-roofline driven):
//  * candidates = 256 MB f32 -> stream ONCE from HBM (~11us @ 23.3 TB/s).
//  * all 256 queries live in LDS as f16 (32 KB) so every candidate tile is
//    reused 16x (16 query tiles) -> HBM traffic stays at 1x candidates.
//  * V_WMMA_F32_16X16X32_F16, K=64 -> 2 WMMAs per 16x16 tile, f32 accum.
//  * fused top-k: per-query threshold (from a WMMA-scored 8192-candidate
//    sample, 16th largest) + atomic capture list (cap 6144/query), then an
//    exact per-query top-(k+E) select, exclusion penalty, final top-k.
// ---------------------------------------------------------------------------

typedef __attribute__((ext_vector_type(16))) _Float16 v16h;
typedef __attribute__((ext_vector_type(8)))  float    v8f;
typedef __attribute__((ext_vector_type(4)))  float    v4f;

#define WPB      8            // waves per block (256 threads, wave32)
#define TPW      8            // candidate tiles per wave in main kernel
#define CAP      6144         // capture capacity per query
#define SS       8192         // sampled candidates for threshold estimate
#define RRELAX   16           // threshold = 16th largest sample
#define NEG_INF  (-3.0e38f)
#define PENALTY  100000.0f

// D = A(16x32 f16) * B(32x16 f16) + C(16x16 f32), wave32, per ISA 7.12.2
__device__ __forceinline__ v8f wmma_f16(v16h a, v16h b, v8f c) {
  return __builtin_amdgcn_wmma_f32_16x16x32_f16(false, a, false, b,
                                                (short)0, c, false, false);
}

// Load 16 contiguous f32 (64B-aligned) and convert to f16 fragment half.
// B-matrix layout: lane (col = lane&15) holds K = 32*kh + 16*(lane>>4) + j.
__device__ __forceinline__ v16h cvt16(const float* __restrict__ p) {
  const v4f* vp = (const v4f*)p;
  v4f x0 = vp[0], x1 = vp[1], x2 = vp[2], x3 = vp[3];
  v16h r;
#pragma unroll
  for (int j = 0; j < 4; ++j) r[j]      = (_Float16)x0[j];
#pragma unroll
  for (int j = 0; j < 4; ++j) r[4 + j]  = (_Float16)x1[j];
#pragma unroll
  for (int j = 0; j < 4; ++j) r[8 + j]  = (_Float16)x2[j];
#pragma unroll
  for (int j = 0; j < 4; ++j) r[12 + j] = (_Float16)x3[j];
  return r;
}

// A-matrix fragment from an LDS f16 query row, K-window [kh*32, kh*32+32):
// halves 0..7  -> K = kh*32 + g*16 + j
// halves 8..15 -> K = kh*32 + g*16 + 8 + j     (matches B contiguous K order)
__device__ __forceinline__ v16h ldsA(const _Float16* __restrict__ row,
                                     int g, int kh) {
  const _Float16* p = row + kh * 32;
  v16h r;
#pragma unroll
  for (int j = 0; j < 8; ++j) r[j]     = p[g * 16 + j];
#pragma unroll
  for (int j = 0; j < 8; ++j) r[8 + j] = p[g * 16 + 8 + j];
  return r;
}

// ---------------------------------------------------------------------------
// Kernel 1: WMMA-score a strided sample of candidates.
// Output layout is TRANSPOSED: sampT[sample][query] so each lane's 8 scores
// (consecutive q) store as two global_store_b128 from the accumulator.
// grid.x = SS/16/WPB blocks of 256 threads; one candidate tile per wave.
// ---------------------------------------------------------------------------
__global__ __launch_bounds__(256)
void k_sample(const float* __restrict__ Q, const float* __restrict__ C,
              float* __restrict__ sampT, int stride, int N, int numQT,
              int B) {
  __shared__ _Float16 qs[256 * 64];
  const int nq = numQT * 16 * 64;
  for (int i = threadIdx.x; i < 256 * 64; i += 256)
    qs[i] = (i < nq) ? (_Float16)Q[i] : (_Float16)0.0f;
  __syncthreads();

  const int lane = threadIdx.x & 31;
  const int wave = threadIdx.x >> 5;
  const int g = lane >> 4, n = lane & 15;
  const int tile = blockIdx.x * WPB + wave;     // 0..SS/16-1, wave-uniform
  const int sidx = tile * 16 + n;
  long long cand = (long long)sidx * stride;
  if (cand >= N) cand = N - 1;                  // clamp (duplicates OK)
  const float* cp = C + cand * 64;

  v16h b0 = cvt16(cp + g * 16);
  v16h b1 = cvt16(cp + 32 + g * 16);

  float* op = sampT + (size_t)sidx * B + g * 8; // per-lane output base

  for (int qt = 0; qt < numQT; ++qt) {
    const _Float16* row = qs + (qt * 16 + n) * 64;
    v16h a0 = ldsA(row, g, 0);
    v16h a1 = ldsA(row, g, 1);
    v8f c = {};
    c = wmma_f16(a0, b0, c);
    c = wmma_f16(a1, b1, c);
    v4f lo, hi;
#pragma unroll
    for (int j = 0; j < 4; ++j) { lo[j] = c[j]; hi[j] = c[4 + j]; }
    *(v4f*)(op + qt * 16)     = lo;             // q = qt*16 + g*8 + 0..3
    *(v4f*)(op + qt * 16 + 4) = hi;             // q = qt*16 + g*8 + 4..7
  }
}

// ---------------------------------------------------------------------------
// Kernel 2: per-query threshold = (RRELAX-th largest sample) - eps.
// Also zeroes the capture counters (replay-safe re-init).
// ---------------------------------------------------------------------------
__global__ __launch_bounds__(256)
void k_thresh(const float* __restrict__ sampT, float* __restrict__ thr,
              int* __restrict__ count, int B) {
  const int q = blockIdx.x, tid = threadIdx.x;
  __shared__ float ls[SS];
  __shared__ float rv[256];
  __shared__ int   ri[256];
  __shared__ float fin;
  for (int i = tid; i < SS; i += 256) ls[i] = sampT[(size_t)i * B + q];
  __syncthreads();
  for (int it = 0; it < RRELAX; ++it) {
    float lv = NEG_INF; int li = -1;
    for (int i = tid; i < SS; i += 256) {
      float v = ls[i];
      if (v > lv) { lv = v; li = i; }
    }
    rv[tid] = lv; ri[tid] = li;
    __syncthreads();
    for (int s = 128; s > 0; s >>= 1) {
      if (tid < s && rv[tid + s] > rv[tid]) {
        rv[tid] = rv[tid + s]; ri[tid] = ri[tid + s];
      }
      __syncthreads();
    }
    if (tid == 0) { fin = rv[0]; if (ri[0] >= 0) ls[ri[0]] = NEG_INF; }
    __syncthreads();
  }
  if (tid == 0) { thr[q] = fin - 1e-3f; count[q] = 0; }
}

// ---------------------------------------------------------------------------
// Kernel 3: main streaming WMMA GEMM + threshold filter + atomic capture.
// Each block: queries (f16) + thresholds staged in LDS; 8 waves x TPW tiles;
// per candidate tile: 16 query tiles x 2 WMMAs; thresholds fetched as two
// ds_load_b128 vectors per (tile, qtile); sparse atomic capture.
// ---------------------------------------------------------------------------
__global__ __launch_bounds__(256)
void k_filter(const float* __restrict__ Q, const float* __restrict__ C,
              const float* __restrict__ thr, int* __restrict__ count,
              float* __restrict__ capS, int* __restrict__ capI,
              int nTiles, int numQT) {
  __shared__ _Float16 qs[256 * 64];
  __shared__ float    sthr[256];
  const int nq = numQT * 16 * 64;
  for (int i = threadIdx.x; i < 256 * 64; i += 256)
    qs[i] = (i < nq) ? (_Float16)Q[i] : (_Float16)0.0f;
  if (threadIdx.x < 256)
    sthr[threadIdx.x] =
        (threadIdx.x < numQT * 16) ? thr[threadIdx.x] : 3.0e38f;
  __syncthreads();

  const int lane = threadIdx.x & 31;
  const int wave = threadIdx.x >> 5;
  const int g = lane >> 4, n = lane & 15;
  const int tileBase = blockIdx.x * (WPB * TPW);

  for (int i = 0; i < TPW; ++i) {
    const int tile = tileBase + i * WPB + wave;     // wave-uniform
    if (tile >= nTiles) continue;
    const int cand = (tile << 4) + n;
    const float* cp = C + (size_t)cand * 64;

    // prefetch next tile's row into cache (global_prefetch_b8)
    if (i + 1 < TPW) {
      const int nt = tileBase + (i + 1) * WPB + wave;
      if (nt < nTiles)
        __builtin_prefetch(C + (size_t)((nt << 4) + n) * 64, 0, 1);
    }

    v16h b0 = cvt16(cp + g * 16);
    v16h b1 = cvt16(cp + 32 + g * 16);

    for (int qt = 0; qt < numQT; ++qt) {
      const _Float16* row = qs + (qt * 16 + n) * 64;
      v16h a0 = ldsA(row, g, 0);
      v16h a1 = ldsA(row, g, 1);
      v8f c = {};
      c = wmma_f16(a0, b0, c);
      c = wmma_f16(a1, b1, c);
      // 8 thresholds for this lane are contiguous -> two b128 LDS loads
      const v4f* tp = (const v4f*)(sthr + qt * 16 + g * 8);
      v4f t0 = tp[0], t1 = tp[1];
#pragma unroll
      for (int r = 0; r < 8; ++r) {
        const float t = (r < 4) ? t0[r] : t1[r - 4];
        const float s = c[r];
        if (s > t) {
          const int q = qt * 16 + g * 8 + r;
          int p = atomicAdd(&count[q], 1);
          if (p < CAP) {
            capS[(size_t)q * CAP + p] = s;
            capI[(size_t)q * CAP + p] = cand;
          }
        }
      }
    }
  }
}

// ---------------------------------------------------------------------------
// Kernel 4: per-query exact top-(k+E) from captures, exclusion penalty,
// final top-k -> out scores then ids (as float), matching reference order.
// ---------------------------------------------------------------------------
__global__ __launch_bounds__(256)
void k_select(const float* __restrict__ capS, const int* __restrict__ capI,
              const int* __restrict__ count, const int* __restrict__ ident,
              const int* __restrict__ excl, float* __restrict__ outS,
              float* __restrict__ outI, int k, int E) {
  const int q = blockIdx.x, tid = threadIdx.x;
  const int ak = k + E;                         // adjusted_k (<=256 assumed)
  __shared__ float ss[CAP];
  __shared__ float rv[256];
  __shared__ int   ri[256];
  __shared__ float topS[256];
  __shared__ int   topId[256];
  __shared__ float adj[256];
  __shared__ int   ex[256];

  int cnt = count[q]; if (cnt > CAP) cnt = CAP;
  for (int i = tid; i < cnt; i += 256) ss[i] = capS[(size_t)q * CAP + i];
  const int Ec = (E < 256) ? E : 256;
  for (int j = tid; j < Ec; j += 256) ex[j] = excl[(size_t)q * E + j];
  __syncthreads();

  // exact top-(k+E), descending
  for (int it = 0; it < ak; ++it) {
    float lv = NEG_INF; int li = -1;
    for (int i = tid; i < cnt; i += 256) {
      float v = ss[i];
      if (v > lv) { lv = v; li = i; }
    }
    rv[tid] = lv; ri[tid] = li;
    __syncthreads();
    for (int s = 128; s > 0; s >>= 1) {
      if (tid < s && rv[tid + s] > rv[tid]) {
        rv[tid] = rv[tid + s]; ri[tid] = ri[tid + s];
      }
      __syncthreads();
    }
    if (tid == 0) {
      const int idx = ri[0];
      topS[it] = rv[0];
      topId[it] = (idx >= 0) ? ident[capI[(size_t)q * CAP + idx]] : -1;
      if (idx >= 0) ss[idx] = NEG_INF;
    }
    __syncthreads();
  }

  // exclusion penalty on the over-fetched list
  for (int i = tid; i < ak; i += 256) {
    const int id = topId[i];
    bool hit = false;
    for (int j = 0; j < Ec; ++j) hit = hit || (id == ex[j]);
    adj[i] = topS[i] - (hit ? PENALTY : 0.0f);
  }
  __syncthreads();

  // final top-k over adjusted scores; emit ORIGINAL score + id
  for (int it = 0; it < k; ++it) {
    float lv = NEG_INF; int li = -1;
    for (int i = tid; i < ak; i += 256) {
      float v = adj[i];
      if (v > lv) { lv = v; li = i; }
    }
    rv[tid] = lv; ri[tid] = li;
    __syncthreads();
    for (int s = 128; s > 0; s >>= 1) {
      if (tid < s && rv[tid + s] > rv[tid]) {
        rv[tid] = rv[tid + s]; ri[tid] = ri[tid + s];
      }
      __syncthreads();
    }
    if (tid == 0) {
      const int idx = ri[0];
      outS[(size_t)q * k + it] = (idx >= 0) ? topS[idx] : NEG_INF;
      outI[(size_t)q * k + it] = (idx >= 0) ? (float)topId[idx] : -1.0f;
      if (idx >= 0) adj[idx] = NEG_INF;
    }
    __syncthreads();
  }
}

// ---------------------------------------------------------------------------
extern "C" void kernel_launch(void* const* d_in, const int* in_sizes, int n_in,
                              void* d_out, int out_size, void* d_ws,
                              size_t ws_size, hipStream_t stream) {
  const float* Q     = (const float*)d_in[0];   // [B, 64] f32
  const float* C     = (const float*)d_in[1];   // [N, 64] f32
  const int*   ident = (const int*)d_in[2];     // [N] i32
  const int*   excl  = (const int*)d_in[3];     // [B, E] i32

  const int D = 64;
  const int B = in_sizes[0] / D;                // 256
  const int N = in_sizes[1] / D;                // 1,000,000
  const int E = in_sizes[3] / B;                // 100
  const int k = out_size / (2 * B);             // 100
  const int numQT = B / 16;                     // 16 query tiles

  // workspace layout (capS aliases sampT: sample phase finishes first)
  char*  w     = (char*)d_ws;
  float* thr   = (float*)w;                       // B floats
  int*   cnt   = (int*)(w + 1024);                // B ints
  float* sampT = (float*)(w + 4096);              // SS*B floats [sample][q]
  float* capS  = sampT;                           // B*CAP floats (alias)
  int*   capI  = (int*)(w + 4096 + (size_t)B * SS * sizeof(float));
  const size_t need = 4096 + (size_t)B * SS * sizeof(float)
                           + (size_t)B * CAP * sizeof(int);
  if (ws_size < need) return;                     // ~14.7 MB required

  float* outS = (float*)d_out;                    // [B, k] scores
  float* outI = outS + (size_t)B * k;             // [B, k] ids (as float)

  int stride = N / SS; if (stride < 1) stride = 1;

  // 1) sample scores (WMMA, transposed dense output)
  k_sample<<<SS / 16 / WPB, 256, 0, stream>>>(Q, C, sampT, stride, N,
                                              numQT, B);
  // 2) thresholds + counter reset
  k_thresh<<<B, 256, 0, stream>>>(sampT, thr, cnt, B);
  // 3) main streaming GEMM + capture
  const int nTiles = N / 16;
  const int per = WPB * TPW;
  k_filter<<<(nTiles + per - 1) / per, 256, 0, stream>>>(
      Q, C, thr, cnt, capS, capI, nTiles, numQT);
  // 4) exact top-k with exclusions
  k_select<<<B, 256, 0, stream>>>(capS, capI, cnt, ident, excl,
                                  outS, outI, k, E);
}
